// GEDNet_57002805952697
// MI455X (gfx1250) — matrix operand order repaired and checked by reading, not compiled, not run on hardware
//
#include <hip/hip_runtime.h>
#include <hip/hip_bf16.h>
#include <math.h>

// ---------------------------------------------------------------------------
// GEDNet forward for MI455X (gfx1250, wave32, WMMA + async-to-LDS).
//
//  * All GEMMs via v_wmma_f32_16x16x32_f16 (f16 in, f32 accumulate).
//  * Weights pre-packed once into fragment-linear f16 (one 32B vector load
//    per B-fragment, no per-use f32->f16 conversion).
//  * Per-pair Sinkhorn runs entirely out of LDS: K is 192x193 f32 = 148 KB,
//    plus f16 embedding tiles (48 KB) => ~194 KB < 320 KB/WGP. HBM is touched
//    once for the 151 MB of outputs (the bandwidth floor at 23.3 TB/s).
//  * Embeddings staged into LDS with GLOBAL_LOAD_ASYNC_TO_LDS_B128
//    (ASYNCcnt path; s_wait_asynccnt fence before the barrier).
//  * K stride 193 (193 mod 64 == 1) => bank-conflict-free column AND row
//    mat-vecs in the Sinkhorn loop.
// ---------------------------------------------------------------------------

#define MAXN   192
#define KSTR   193
#define EMBD   64
#define SINKIT 8

typedef __attribute__((ext_vector_type(16))) _Float16 v16h;
typedef __attribute__((ext_vector_type(8)))  _Float16 v8h;
typedef __attribute__((ext_vector_type(8)))  float    v8f;

__device__ __forceinline__ v8f wmma_f16(v16h a, v16h b, v8f c) {
  // D(16x16 f32) = A(16x32 f16) * B(32x16 f16) + C
  return __builtin_amdgcn_wmma_f32_16x16x32_f16(
      /*neg_a=*/false, a, /*neg_b=*/false, b,
      /*c_mod=*/(short)0, c, /*reuse_a=*/false, /*reuse_b=*/false);
}

// ---- WMMA operand layouts used here (wave32; A/B k order is this code's, not the ISA table order) -----------
// A 16x32 f16: lane m / m+16 -> row m. Element e of the v16h:
//   k = kbase + e + (lane>=16 ? 16 : 0)
// B 32x16 f16: lane n / n+16 -> col n. Element e: k = kbase + e + (lane>=16?16:0)
// C/D 16x16 f32: VGPR r -> row r + (lane>=16 ? 8 : 0), col = lane & 15.

__device__ __forceinline__ v16h a_frag_g32(const float* A, int lda, int row0,
                                           int kb, int N) {
  int lane = threadIdx.x & 31, hf = lane >> 4, m = lane & 15;
  int r = row0 + m; if (r > N - 1) r = N - 1;
  const float* p = A + (long)r * lda + kb + hf * 16;
  v16h a;
#pragma unroll
  for (int e = 0; e < 8; ++e) { a[e] = (_Float16)p[e]; a[8 + e] = (_Float16)p[8 + e]; }
  return a;
}

__device__ __forceinline__ v16h a_frag_g16(const _Float16* A, int lda, int row0,
                                           int kb, int N) {
  int lane = threadIdx.x & 31, hf = lane >> 4, m = lane & 15;
  int r = row0 + m; if (r > N - 1) r = N - 1;
  const _Float16* p = A + (long)r * lda + kb + hf * 16;
  v16h a;
#pragma unroll
  for (int e = 0; e < 8; ++e) { a[e] = p[e]; a[8 + e] = p[8 + e]; }
  return a;
}

// Fragment-linear packed weights: Wp[((k/32)*64 + n)*32 + (k%32)] = W[k][n].
// Per lane the 16 needed halfs are contiguous => one 32B load, no cvt.
__device__ __forceinline__ v16h b_frag_wp(const _Float16* Wp, int kb, int col0) {
  int lane = threadIdx.x & 31, hf = lane >> 4, n = lane & 15;
  return *(const v16h*)(Wp + (((kb >> 5) * 64 + col0 + n) << 5) + hf * 16);
}

// LDS, embeddings row-major [192 x 64] f16, 32B-aligned rows.
__device__ __forceinline__ v16h a_frag_lds(const _Float16* S, int row0, int kb) {
  int lane = threadIdx.x & 31, hf = lane >> 4, m = lane & 15;
  const _Float16* p = S + (row0 + m) * EMBD + kb + hf * 16;
  v8h lo = *(const v8h*)p;
  v8h hi = *(const v8h*)(p + 8);
  v16h a;
#pragma unroll
  for (int e = 0; e < 8; ++e) { a[e] = lo[e]; a[8 + e] = hi[e]; }
  return a;
}

// B = T^T: B[k][n] = T[n][k]  =>  per lane a CONTIGUOUS 16-half (32B) LDS load.
__device__ __forceinline__ v16h bT_frag_lds(const _Float16* T, int col0, int kb) {
  int lane = threadIdx.x & 31, hf = lane >> 4, n = lane & 15;
  return *(const v16h*)(T + (col0 + n) * EMBD + kb + hf * 16);
}

// gfx1250 async copy: per-lane 16B global -> LDS, tracked by ASYNCcnt
// (CDNA5 ISA, CDNA5 ISA opcode 98). EXEC is all-ones
// at every issue site below.
__device__ __forceinline__ void async_g2l_b128(unsigned lds_off, const void* g) {
  asm volatile("global_load_async_to_lds_b128 %0, %1, off"
               :: "v"(lds_off), "v"((unsigned long long)(size_t)g) : "memory");
}
__device__ __forceinline__ void wait_asynccnt0() {
#if __has_builtin(__builtin_amdgcn_s_wait_asynccnt)
  __builtin_amdgcn_s_wait_asynccnt(0);
#else
  asm volatile("s_wait_asynccnt 0" ::: "memory");
#endif
}

// ---------------------------------------------------------------------------
// prefix offsets + normalized virtual node (f16)
__global__ void prep_kernel(const int* len_s, const int* len_t, const float* virt,
                            int* off_s, int* off_t, _Float16* virt_h, int P) {
  int t = threadIdx.x;
  if (t < P) {
    int a = 0, b = 0;
    for (int q = 0; q < t; ++q) { a += len_s[q]; b += len_t[q]; }
    off_s[t] = a; off_t[t] = b;
  }
  if (t == 0) {
    float s = 0.f;
    for (int d = 0; d < 64; ++d) s += virt[d] * virt[d];
    float sc = 1.f / fmaxf(sqrtf(s), 1e-12f);
    for (int d = 0; d < 64; ++d) virt_h[d] = (_Float16)(virt[d] * sc);
  }
}

// pack a [K x 64] f32 weight into fragment-linear f16
__global__ void pack_w_kernel(const float* W, _Float16* Wp, int K) {
  int tot = K * 64;
  for (int t = blockIdx.x * blockDim.x + threadIdx.x; t < tot;
       t += gridDim.x * blockDim.x) {
    int k = t / 64, n = t - (k * 64);
    Wp[(((k >> 5) * 64 + n) << 5) + (k & 31)] = (_Float16)W[t];
  }
}

// segment_sum: agg[dst] += feat[src]
__global__ void scatter_f32(const int* ei, long nE, const float* feat, int C, float* agg) {
  long t = (long)blockIdx.x * blockDim.x + threadIdx.x;
  if (t >= nE * (long)C) return;
  long e = t / C; int c = (int)(t - e * C);
  atomicAdd(&agg[(long)ei[nE + e] * C + c], feat[(long)ei[e] * C + c]);
}
__global__ void scatter_f16(const int* ei, long nE, const _Float16* feat, int C, float* agg) {
  long t = (long)blockIdx.x * blockDim.x + threadIdx.x;
  if (t >= nE * (long)C) return;
  long e = t / C; int c = (int)(t - e * C);
  atomicAdd(&agg[(long)ei[nE + e] * C + c], (float)feat[(long)ei[e] * C + c]);
}

// h_out = relu(agg @ Wr + br + prev @ Wo)   (one GraphConv layer)
__global__ void __launch_bounds__(256)
gnn_layer_kernel(const float* agg, const float* prevf, const _Float16* prevh, int K,
                 const _Float16* Wrp, const float* br, const _Float16* Wop,
                 _Float16* out, int N) {
  int wid = threadIdx.x >> 5;
  int row0 = (blockIdx.x * (blockDim.x >> 5) + wid) * 16;
  if (row0 >= N) return;
  int lane = threadIdx.x & 31, hf = lane >> 4, nn = lane & 15;
  for (int nt = 0; nt < 4; ++nt) {
    int col0 = nt * 16;
    v8f acc = {0.f, 0.f, 0.f, 0.f, 0.f, 0.f, 0.f, 0.f};
    for (int kb = 0; kb < K; kb += 32) {
      acc = wmma_f16(a_frag_g32(agg, K, row0, kb, N), b_frag_wp(Wrp, kb, col0), acc);
      v16h ap = prevh ? a_frag_g16(prevh, K, row0, kb, N)
                      : a_frag_g32(prevf, K, row0, kb, N);
      acc = wmma_f16(ap, b_frag_wp(Wop, kb, col0), acc);
    }
    float bias = br[col0 + nn];
#pragma unroll
    for (int r = 0; r < 8; ++r) {
      int row = row0 + r + hf * 8;
      if (row < N) out[(long)row * 64 + col0 + nn] = (_Float16)fmaxf(acc[r] + bias, 0.f);
    }
  }
}

// emb = normalize([x | h1 | h2] @ W_e + b_e), stored f16
__global__ void __launch_bounds__(256)
emb_kernel(const float* x, const _Float16* h1, const _Float16* h2,
           const _Float16* Wep, const float* be, _Float16* emb, int N) {
  int wid = threadIdx.x >> 5;
  int row0 = (blockIdx.x * (blockDim.x >> 5) + wid) * 16;
  if (row0 >= N) return;
  int lane = threadIdx.x & 31, hf = lane >> 4, nn = lane & 15;
  v8f acc[4];
  for (int nt = 0; nt < 4; ++nt) {
    int col0 = nt * 16;
    v8f a = {0.f, 0.f, 0.f, 0.f, 0.f, 0.f, 0.f, 0.f};
    a = wmma_f16(a_frag_g32(x, 32, row0, 0, N),   b_frag_wp(Wep, 0,   col0), a);
    a = wmma_f16(a_frag_g16(h1, 64, row0, 0, N),  b_frag_wp(Wep, 32,  col0), a);
    a = wmma_f16(a_frag_g16(h1, 64, row0, 32, N), b_frag_wp(Wep, 64,  col0), a);
    a = wmma_f16(a_frag_g16(h2, 64, row0, 0, N),  b_frag_wp(Wep, 96,  col0), a);
    a = wmma_f16(a_frag_g16(h2, 64, row0, 32, N), b_frag_wp(Wep, 128, col0), a);
    float bias = be[col0 + nn];
#pragma unroll
    for (int r = 0; r < 8; ++r) a[r] += bias;
    acc[nt] = a;
  }
  // per-row L2 norm: reduce over 4 col-tiles and the 16 lanes of this half-group
  float sc[8];
#pragma unroll
  for (int r = 0; r < 8; ++r) {
    float t = 0.f;
#pragma unroll
    for (int nt = 0; nt < 4; ++nt) t += acc[nt][r] * acc[nt][r];
#pragma unroll
    for (int msk = 1; msk < 16; msk <<= 1) t += __shfl_xor(t, msk, 32);
    sc[r] = 1.f / fmaxf(sqrtf(t), 1e-12f);
  }
#pragma unroll
  for (int nt = 0; nt < 4; ++nt) {
#pragma unroll
    for (int r = 0; r < 8; ++r) {
      int row = row0 + r + hf * 8;
      if (row < N) emb[(long)row * 64 + nt * 16 + nn] = (_Float16)(acc[nt][r] * sc[r]);
    }
  }
}

// tile dot product S[ti..ti+15] . T[tj..tj+15]^T over d=64 (two WMMAs)
__device__ __forceinline__ v8f tile_dot(const _Float16* Ssh, const _Float16* Tsh,
                                        int ti, int tj) {
  v8f acc = {0.f, 0.f, 0.f, 0.f, 0.f, 0.f, 0.f, 0.f};
  acc = wmma_f16(a_frag_lds(Ssh, ti, 0),  bT_frag_lds(Tsh, tj, 0),  acc);
  acc = wmma_f16(a_frag_lds(Ssh, ti, 32), bT_frag_lds(Tsh, tj, 32), acc);
  return acc;
}

#define PAIR_SMEM (MAXN * KSTR * 4 + 2 * MAXN * EMBD * 2 + (2 * MAXN + 8) * 4)

// One workgroup (8 waves) per graph pair: async-stage embeddings, cost/K build
// (WMMA), 8 Sinkhorn iterations out of LDS, single write of outputs to HBM.
__global__ void __launch_bounds__(256)
pair_kernel(const _Float16* emb_s, const _Float16* emb_t, const _Float16* virt,
            const int* off_s, const int* off_t, const int* len_s, const int* len_t,
            int Ns, int Nt, float* gamma_out, float* cost_out, float* geds_out) {
  extern __shared__ __align__(32) char smem[];
  float*    Ksh = (float*)smem;                               // [192][193] f32
  _Float16* Ssh = (_Float16*)(smem + MAXN * KSTR * 4);        // [192][64]  f16
  _Float16* Tsh = Ssh + MAXN * EMBD;                          // [192][64]  f16
  float*    u   = (float*)(Tsh + MAXN * EMBD);                // [192]
  float*    v   = u + MAXN;                                   // [192]
  float*    red = v + MAXN;

  int p = blockIdx.x, tid = threadIdx.x;
  int n = len_s[p], m = len_t[p];
  int offs = off_s[p], offt = off_t[p];

  // async-stage embeddings into LDS, 16B per lane per issue; exactly 6 full
  // iterations of all 256 threads => EXEC all-ones at every issue.
  for (int q = tid; q < MAXN * 8; q += 256) {
    int row = q >> 3, c = (q & 7) * 8;                        // 8 halfs = 16B
    int sr = offs + row; if (sr > Ns - 1) sr = Ns - 1;
    const _Float16* sp = emb_s + (long)sr * EMBD + c;
    const _Float16* tp = (row < m) ? emb_t + (long)(offt + row) * EMBD + c : virt + c;
    async_g2l_b128((unsigned)(size_t)(Ssh + row * EMBD + c), sp);
    async_g2l_b128((unsigned)(size_t)(Tsh + row * EMBD + c), tp);
  }
  if (tid == 0) red[0] = 0.f;
  if (tid < MAXN) u[tid] = 1.f / (float)MAXN;
  wait_asynccnt0();
  __syncthreads();

  int wid = tid >> 5, lane = tid & 31, hf = lane >> 4, nn = lane & 15;

  // build K = exp(-cost/REG); rows are unit vectors => cd = sqrt(2 - 2*dot)
  for (int t = wid; t < 144; t += 8) {
    int ti = (t / 12) * 16, tj = (t % 12) * 16;
    v8f acc = tile_dot(Ssh, Tsh, ti, tj);
#pragma unroll
    for (int r = 0; r < 8; ++r) {
      int i = ti + r + hf * 8, j = tj + nn;
      float cd = sqrtf(fmaxf(2.f - 2.f * acc[r], 1e-12f));
      float cost = (i < n) ? ((j < n) ? cd : 1000.f) : ((j < n) ? 1000.f : 0.f);
      Ksh[i * KSTR + j] = __expf(-10.f * cost);
    }
  }
  __syncthreads();

  // Sinkhorn: v = 1/(K^T u); u = 1/(K v). Stride 193 => conflict-free both ways.
  for (int it = 0; it < SINKIT; ++it) {
    if (tid < MAXN) {
      float s = 0.f;
      for (int i = 0; i < MAXN; ++i) s += Ksh[i * KSTR + tid] * u[i];
      v[tid] = 1.f / s;
    }
    __syncthreads();
    if (tid < MAXN) {
      float s = 0.f;
      for (int j = 0; j < MAXN; ++j) s += Ksh[tid * KSTR + j] * v[j];
      u[tid] = 1.f / s;
    }
    __syncthreads();
  }

  // outputs: gamma = u K v, cost (recomputed via a 2nd cheap WMMA pass), geds2
  long base = (long)p * (MAXN * MAXN);
  float part = 0.f;
  for (int t = wid; t < 144; t += 8) {
    int ti = (t / 12) * 16, tj = (t % 12) * 16;
    v8f acc = tile_dot(Ssh, Tsh, ti, tj);
#pragma unroll
    for (int r = 0; r < 8; ++r) {
      int i = ti + r + hf * 8, j = tj + nn;
      float cd = sqrtf(fmaxf(2.f - 2.f * acc[r], 1e-12f));
      float cost = (i < n) ? ((j < n) ? cd : 1000.f) : ((j < n) ? 1000.f : 0.f);
      float g = u[i] * Ksh[i * KSTR + j] * v[j];
      gamma_out[base + (long)i * MAXN + j] = g;
      cost_out [base + (long)i * MAXN + j] = cost;
      part += g * cost;
    }
  }
  atomicAdd(red, part);
  __syncthreads();
  if (tid == 0) geds_out[p] = red[0] / (float)(n + m);
}

// ---------------------------------------------------------------------------
extern "C" void kernel_launch(void* const* d_in, const int* in_sizes, int n_in,
                              void* d_out, int out_size, void* d_ws, size_t ws_size,
                              hipStream_t stream) {
  const float* x_s     = (const float*)d_in[0];
  const float* x_t     = (const float*)d_in[1];
  const float* W_rel0  = (const float*)d_in[2];
  const float* b_rel0  = (const float*)d_in[3];
  const float* W_root0 = (const float*)d_in[4];
  const float* W_rel1  = (const float*)d_in[5];
  const float* b_rel1  = (const float*)d_in[6];
  const float* W_root1 = (const float*)d_in[7];
  const float* W_e     = (const float*)d_in[8];
  const float* b_e     = (const float*)d_in[9];
  const float* virt    = (const float*)d_in[10];
  const int*   ei_s    = (const int*)d_in[11];
  const int*   ei_t    = (const int*)d_in[12];
  const int*   len_s   = (const int*)d_in[13];
  const int*   len_t   = (const int*)d_in[14];

  int  Ns  = in_sizes[0] / 32;
  int  Nt  = in_sizes[1] / 32;
  long nEs = in_sizes[11] / 2;
  long nEt = in_sizes[12] / 2;
  int  P   = in_sizes[13];

  // workspace carve (all buffers fully overwritten each call)
  char* w = (char*)d_ws;
  auto carve = [&](size_t bytes) -> char* {
    char* r = w; w += (bytes + 255) & ~(size_t)255; return r;
  };
  float*    agg_s = (float*)   carve((size_t)Ns * 64 * 4);
  _Float16* h1_s  = (_Float16*)carve((size_t)Ns * 64 * 2);
  _Float16* h2_s  = (_Float16*)carve((size_t)Ns * 64 * 2);
  _Float16* es    = (_Float16*)carve((size_t)Ns * 64 * 2);
  float*    agg_t = (float*)   carve((size_t)Nt * 64 * 4);
  _Float16* h1_t  = (_Float16*)carve((size_t)Nt * 64 * 2);
  _Float16* h2_t  = (_Float16*)carve((size_t)Nt * 64 * 2);
  _Float16* et    = (_Float16*)carve((size_t)Nt * 64 * 2);
  int*      offs  = (int*)     carve((size_t)P * 4);
  int*      offt  = (int*)     carve((size_t)P * 4);
  _Float16* vh    = (_Float16*)carve(128);
  _Float16* wp_rel0  = (_Float16*)carve((size_t)32  * 64 * 2);
  _Float16* wp_root0 = (_Float16*)carve((size_t)32  * 64 * 2);
  _Float16* wp_rel1  = (_Float16*)carve((size_t)64  * 64 * 2);
  _Float16* wp_root1 = (_Float16*)carve((size_t)64  * 64 * 2);
  _Float16* wp_e     = (_Float16*)carve((size_t)160 * 64 * 2);

  prep_kernel<<<1, 512, 0, stream>>>(len_s, len_t, virt, offs, offt, vh, P);
  pack_w_kernel<<<8,  256, 0, stream>>>(W_rel0,  wp_rel0,  32);
  pack_w_kernel<<<8,  256, 0, stream>>>(W_root0, wp_root0, 32);
  pack_w_kernel<<<16, 256, 0, stream>>>(W_rel1,  wp_rel1,  64);
  pack_w_kernel<<<16, 256, 0, stream>>>(W_root1, wp_root1, 64);
  pack_w_kernel<<<40, 256, 0, stream>>>(W_e,     wp_e,     160);

  for (int g = 0; g < 2; ++g) {
    const float* x    = g ? x_t   : x_s;
    const int*   ei   = g ? ei_t  : ei_s;
    long         nE   = g ? nEt   : nEs;
    int          N    = g ? Nt    : Ns;
    float*       agg  = g ? agg_t : agg_s;
    _Float16*    h1   = g ? h1_t  : h1_s;
    _Float16*    h2   = g ? h2_t  : h2_s;
    _Float16*    emb  = g ? et    : es;
    int nblk = (N + 127) / 128;

    hipMemsetAsync(agg, 0, (size_t)N * 32 * 4, stream);
    scatter_f32<<<(int)((nE * 32 + 255) / 256), 256, 0, stream>>>(ei, nE, x, 32, agg);
    gnn_layer_kernel<<<nblk, 256, 0, stream>>>(agg, x, nullptr, 32,
                                               wp_rel0, b_rel0, wp_root0, h1, N);
    hipMemsetAsync(agg, 0, (size_t)N * 64 * 4, stream);
    scatter_f16<<<(int)((nE * 64 + 255) / 256), 256, 0, stream>>>(ei, nE, h1, 64, agg);
    gnn_layer_kernel<<<nblk, 256, 0, stream>>>(agg, nullptr, h1, 64,
                                               wp_rel1, b_rel1, wp_root1, h2, N);
    emb_kernel<<<nblk, 256, 0, stream>>>(x, h1, h2, wp_e, b_e, emb, N);
  }

  float* gamma = (float*)d_out;
  float* cost  = gamma + (long)P * MAXN * MAXN;
  float* geds  = cost  + (long)P * MAXN * MAXN;

  hipFuncSetAttribute(reinterpret_cast<const void*>(&pair_kernel),
                      hipFuncAttributeMaxDynamicSharedMemorySize, PAIR_SMEM);
  pair_kernel<<<P, 256, PAIR_SMEM, stream>>>(es, et, vh, offs, offt, len_s, len_t,
                                             Ns, Nt, gamma, cost, geds);
}
